// StandardMoELayer_53068615910180
// MI455X (gfx1250) — compile-verified, matrix operand order checked by reading, not run on hardware
//
#include <hip/hip_runtime.h>
#include <hip/hip_bf16.h>
#include <math.h>

typedef __bf16 bf16;
typedef __attribute__((ext_vector_type(16))) __bf16 v16bf;
typedef __attribute__((ext_vector_type(8)))  float  v8f;
typedef __attribute__((ext_vector_type(4)))  unsigned int v4u;

#define T_TOK   16384      // B*S = 4*4096
#define D_MODEL 1024
#define N_EXP   16
#define D_FFN   32
#define N_FFN   512        // N_EXP * D_FFN
#define S_SEQ   4096

// Use CDNA5 async global->LDS copies (ASYNCcnt) for GEMM staging.
// Flip to 0 to fall back to register staging if the assembler rejects it.
#define USE_ASYNC_LDS 1

// ---- workspace layout (bytes) ----
static constexpr size_t OFF_SUMS  = 0;
static constexpr size_t OFF_SUMSQ = 64;
static constexpr size_t OFF_USAGE = 128;
static constexpr size_t OFF_W     = 256;                                   // combine weights [T,16] f32
static constexpr size_t OFF_XBF   = OFF_W   + (size_t)T_TOK * 16 * 4;      // x bf16 [T,1024]
static constexpr size_t OFF_W1T   = OFF_XBF + (size_t)T_TOK * D_MODEL * 2; // W1t bf16 [512][1024] (K-contig)
static constexpr size_t OFF_W2T   = OFF_W1T + (size_t)N_FFN * D_MODEL * 2; // W2t bf16 [1024][512] (K-contig)
static constexpr size_t OFF_H     = OFF_W2T + (size_t)D_MODEL * N_FFN * 2; // h bf16 [T,512]

union FragU { v4u q[2]; v16bf v; };

// -------------------- init --------------------
__global__ void init_kernel(float* sums, float* sumsq, float* usage) {
    int i = threadIdx.x;
    if (i < 4)  { sums[i] = 0.f; sumsq[i] = 0.f; }
    if (i < 16) usage[i] = 0.f;
}

// -------------------- batch LN stats --------------------
__global__ __launch_bounds__(256)
void stats_kernel(const float* __restrict__ x, float* sums, float* sumsq) {
    const int b = blockIdx.y;
    const size_t n = (size_t)S_SEQ * D_MODEL;
    const float* xb = x + (size_t)b * n;
    float s = 0.f, s2 = 0.f;
    for (size_t i = (size_t)blockIdx.x * blockDim.x + threadIdx.x; i < n;
         i += (size_t)gridDim.x * blockDim.x) {
        float v = xb[i]; s += v; s2 += v * v;
    }
    __shared__ float rs[256], rq[256];
    int tid = threadIdx.x;
    rs[tid] = s; rq[tid] = s2;
    __syncthreads();
    for (int off = 128; off > 0; off >>= 1) {
        if (tid < off) { rs[tid] += rs[tid + off]; rq[tid] += rq[tid + off]; }
        __syncthreads();
    }
    if (tid == 0) { atomicAdd(&sums[b], rs[0]); atomicAdd(&sumsq[b], rq[0]); }
}

// -------------------- bf16 conversions --------------------
__global__ void cvt_x_kernel(const float* __restrict__ x, bf16* __restrict__ xb, size_t n) {
    for (size_t i = (size_t)blockIdx.x * blockDim.x + threadIdx.x; i < n;
         i += (size_t)gridDim.x * blockDim.x)
        xb[i] = (bf16)x[i];
}
// W1[e][d][f] -> W1t[e*32+f][d]
__global__ void cvt_w1_kernel(const float* __restrict__ W1, bf16* __restrict__ W1t) {
    int i = blockIdx.x * blockDim.x + threadIdx.x;          // over 512*1024
    if (i < N_FFN * D_MODEL) {
        int n = i >> 10, d = i & 1023;
        int e = n >> 5, f = n & 31;
        W1t[i] = (bf16)W1[(size_t)e * D_MODEL * D_FFN + (size_t)d * D_FFN + f];
    }
}
// W2[ef][d] -> W2t[d][ef]
__global__ void cvt_w2_kernel(const float* __restrict__ W2, bf16* __restrict__ W2t) {
    int i = blockIdx.x * blockDim.x + threadIdx.x;          // over 1024*512
    if (i < D_MODEL * N_FFN) {
        int d = i >> 9, ef = i & 511;
        W2t[i] = (bf16)W2[(size_t)ef * D_MODEL + d];
    }
}

// -------------------- gating: logits, top-2 softmax, usage --------------------
__global__ __launch_bounds__(256)
void gate_kernel(const float* __restrict__ x, const int* __restrict__ task_id,
                 const float* __restrict__ task_emb, const float* __restrict__ Wg,
                 const float* __restrict__ bg, const float* __restrict__ sums,
                 const float* __restrict__ sumsq, float* __restrict__ wcomb,
                 float* __restrict__ usage, float* __restrict__ topk_out) {
    __shared__ float lg[8][32];
    const int tid = threadIdx.x, lane = tid & 31, wave = tid >> 5;
    const int t = (blockIdx.x * 8 + wave) * 2 + (lane >> 4);   // token
    const int e = lane & 15;                                    // expert
    const int b = t >> 12;                                      // S = 4096
    const float invN = 1.f / ((float)S_SEQ * (float)D_MODEL);
    const float mu   = sums[b] * invN;
    const float var  = sumsq[b] * invN - mu * mu;
    const float rstd = rsqrtf(var + 1e-5f);

    const float* xr = x + (size_t)t * D_MODEL;
    float acc = bg[e];
    #pragma unroll 4
    for (int d = 0; d < D_MODEL; ++d)
        acc += (xr[d] - mu) * rstd * Wg[d * N_EXP + e];
    const int tb = task_id[b];
    const float* te = task_emb + (size_t)tb * 64;
    #pragma unroll 4
    for (int j = 0; j < 64; ++j)
        acc += te[j] * Wg[(D_MODEL + j) * N_EXP + e];

    lg[wave][lane] = acc;
    __syncthreads();

    if (e == 0) {
        const float* L = &lg[wave][(lane >> 4) * 16];
        int i0 = 0; float v0 = L[0];
        for (int k = 1; k < 16; ++k) if (L[k] > v0) { v0 = L[k]; i0 = k; }
        int i1 = (i0 == 0) ? 1 : 0; float v1 = L[i1];
        for (int k = 0; k < 16; ++k)
            if (k != i0 && L[k] > v1) { v1 = L[k]; i1 = k; }
        float mx = fmaxf(v0, v1);
        float e0 = __expf(v0 - mx), e1 = __expf(v1 - mx);
        float inv = 1.f / (e0 + e1);
        float* wr = wcomb + (size_t)t * N_EXP;
        #pragma unroll
        for (int k = 0; k < 16; ++k) wr[k] = 0.f;
        wr[i0] = e0 * inv; wr[i1] = e1 * inv;
        topk_out[(size_t)t * 2]     = (float)i0;
        topk_out[(size_t)t * 2 + 1] = (float)i1;
        atomicAdd(&usage[i0], 1.f);
        atomicAdd(&usage[i1], 1.f);
    }
}

// -------------------- WMMA GEMM (bf16 in, f32 acc) --------------------
// C[M,N] = A[M,K] @ Bt[N,K]^T ; EPI=1: silu(+b1)*w -> bf16 h ; EPI=2: + w@b2 -> f32 out
// Block tile 128x128xBK32; 8 waves (4x2); wave tile 32x64 = 8 WMMAs per K-step.
#if USE_ASYNC_LDS
__device__ __forceinline__ void async_copy_b128(unsigned ldsAddr, const bf16* g) {
    asm volatile("global_load_async_to_lds_b128 %0, %1, off"
                 :: "v"(ldsAddr), "v"((unsigned long long)(uintptr_t)g)
                 : "memory");
}
#endif

template <int EPI>
__global__ __launch_bounds__(256)
void moe_gemm_kernel(const bf16* __restrict__ A, const bf16* __restrict__ Bt,
                     const float* __restrict__ wcomb, const float* __restrict__ bias,
                     bf16* __restrict__ Hout, float* __restrict__ Fout,
                     int M, int N, int K) {
    constexpr int BM = 128, BN = 128, BK = 32;
    __shared__ __align__(16) bf16 As[2][BM * BK];
    __shared__ __align__(16) bf16 Bs[2][BN * BK];

    const int tid  = threadIdx.x;
    const int lane = tid & 31;
    const int wave = tid >> 5;
    const int wm = wave >> 1, wn = wave & 1;    // 4x2 wave grid, wave tile 32x64
    const int m0 = blockIdx.y * BM;
    const int n0 = blockIdx.x * BN;

    const v8f zero = {0.f, 0.f, 0.f, 0.f, 0.f, 0.f, 0.f, 0.f};
    v8f acc[2][4];
    #pragma unroll
    for (int i = 0; i < 2; ++i)
        #pragma unroll
        for (int j = 0; j < 4; ++j) acc[i][j] = zero;

    const int rsel = lane & 15;
    const int hk   = (lane >> 4) << 3;          // 0 or 8 (16-bit A/B fragment layout)

    // per-thread staging coordinates: 512 x b128 per tile, 2 per thread
    const int sRow0 = tid >> 2, sKq = tid & 3;          // idx = tid
    const int sRow1 = (tid + 256) >> 2;                 // idx = tid + 256

#if USE_ASYNC_LDS
    auto issue_tile = [&](int buf, int k0) {
        async_copy_b128((unsigned)(uintptr_t)&As[buf][sRow0 * BK + sKq * 8],
                        &A[(size_t)(m0 + sRow0) * K + k0 + sKq * 8]);
        async_copy_b128((unsigned)(uintptr_t)&As[buf][sRow1 * BK + sKq * 8],
                        &A[(size_t)(m0 + sRow1) * K + k0 + sKq * 8]);
        async_copy_b128((unsigned)(uintptr_t)&Bs[buf][sRow0 * BK + sKq * 8],
                        &Bt[(size_t)(n0 + sRow0) * K + k0 + sKq * 8]);
        async_copy_b128((unsigned)(uintptr_t)&Bs[buf][sRow1 * BK + sKq * 8],
                        &Bt[(size_t)(n0 + sRow1) * K + k0 + sKq * 8]);
    };
    issue_tile(0, 0);
#endif

    int cur = 0;
    for (int k0 = 0; k0 < K; k0 += BK) {
#if USE_ASYNC_LDS
        if (k0 + BK < K) {
            issue_tile(cur ^ 1, k0 + BK);                 // prefetch next tile
            asm volatile("s_wait_asynccnt 4" ::: "memory"); // tile k landed in LDS
        } else {
            asm volatile("s_wait_asynccnt 0" ::: "memory");
        }
        __syncthreads();
#else
        {
            *(v4u*)&As[cur][sRow0 * BK + sKq * 8] =
                *(const v4u*)&A[(size_t)(m0 + sRow0) * K + k0 + sKq * 8];
            *(v4u*)&As[cur][sRow1 * BK + sKq * 8] =
                *(const v4u*)&A[(size_t)(m0 + sRow1) * K + k0 + sKq * 8];
            *(v4u*)&Bs[cur][sRow0 * BK + sKq * 8] =
                *(const v4u*)&Bt[(size_t)(n0 + sRow0) * K + k0 + sKq * 8];
            *(v4u*)&Bs[cur][sRow1 * BK + sKq * 8] =
                *(const v4u*)&Bt[(size_t)(n0 + sRow1) * K + k0 + sKq * 8];
        }
        __syncthreads();
#endif

        FragU af[2], bf[4];
        #pragma unroll
        for (int i = 0; i < 2; ++i) {
            const bf16* pa = &As[cur][(wm * 32 + i * 16 + rsel) * BK + hk];
            af[i].q[0] = *(const v4u*)pa;
            af[i].q[1] = *(const v4u*)(pa + 16);
        }
        #pragma unroll
        for (int j = 0; j < 4; ++j) {
            const bf16* pb = &Bs[cur][(wn * 64 + j * 16 + rsel) * BK + hk];
            bf[j].q[0] = *(const v4u*)pb;
            bf[j].q[1] = *(const v4u*)(pb + 16);
        }
        #pragma unroll
        for (int i = 0; i < 2; ++i)
            #pragma unroll
            for (int j = 0; j < 4; ++j)
                acc[i][j] = __builtin_amdgcn_wmma_f32_16x16x32_bf16(
                    false, af[i].v, false, bf[j].v, (short)0, acc[i][j], false, false);
        __syncthreads();
        cur ^= 1;
    }

    // epilogue: C layout -> VGPR r: M = r + 8*(lane>>4); N = lane&15
    const int mBase = m0 + wm * 32;
    const int nBase = n0 + wn * 64;
    const int lrow  = (lane >> 4) * 8;
    const int lcol  = lane & 15;

    if (EPI == 1) {
        #pragma unroll
        for (int i = 0; i < 2; ++i)
            #pragma unroll
            for (int j = 0; j < 4; ++j) {
                int n = nBase + j * 16 + lcol;
                int e = n >> 5;
                float b1v = bias[n];                       // b1 flat [512]
                #pragma unroll
                for (int r = 0; r < 8; ++r) {
                    int m = mBase + i * 16 + lrow + r;
                    float c = acc[i][j][r] + b1v;
                    float s = c / (1.f + __expf(-c));      // silu
                    Hout[(size_t)m * N + n] = (bf16)(s * wcomb[(size_t)m * N_EXP + e]);
                }
            }
    } else {
        #pragma unroll
        for (int j = 0; j < 4; ++j) {
            int n = nBase + j * 16 + lcol;
            float b2c[16];
            #pragma unroll
            for (int e = 0; e < 16; ++e) b2c[e] = bias[(size_t)e * N + n];  // b2 [16,1024]
            #pragma unroll
            for (int i = 0; i < 2; ++i)
                #pragma unroll
                for (int r = 0; r < 8; ++r) {
                    int m = mBase + i * 16 + lrow + r;
                    const float* wr = &wcomb[(size_t)m * N_EXP];
                    float bv = 0.f;
                    #pragma unroll
                    for (int e = 0; e < 16; ++e) bv += b2c[e] * wr[e];
                    Fout[(size_t)m * N + n] = acc[i][j][r] + bv;
                }
        }
    }
}

// -------------------- load-balance loss --------------------
__global__ void loss_kernel(const float* __restrict__ usage, float* __restrict__ out_loss) {
    if (threadIdx.x == 0) {
        float s = 0.f;
        for (int e = 0; e < 16; ++e) s += usage[e];
        float mean = s * (1.f / 16.f);
        float ss = 0.f;
        for (int e = 0; e < 16; ++e) { float d = usage[e] - mean; ss += d * d; }
        float stdv = sqrtf(ss * (1.f / 15.f));      // unbiased (ddof=1)
        float m2 = mean + 1e-6f;
        float r = stdv / m2;
        out_loss[0] = r * r;
    }
}

extern "C" void kernel_launch(void* const* d_in, const int* in_sizes, int n_in,
                              void* d_out, int out_size, void* d_ws, size_t ws_size,
                              hipStream_t stream) {
    const float* x        = (const float*)d_in[0];
    const int*   task_id  = (const int*)d_in[1];
    const float* task_emb = (const float*)d_in[2];
    const float* Wg       = (const float*)d_in[3];
    const float* bg       = (const float*)d_in[4];
    const float* W1       = (const float*)d_in[5];
    const float* b1       = (const float*)d_in[6];
    const float* W2       = (const float*)d_in[7];
    const float* b2       = (const float*)d_in[8];

    char* ws = (char*)d_ws;
    float* sums  = (float*)(ws + OFF_SUMS);
    float* sumsq = (float*)(ws + OFF_SUMSQ);
    float* usage = (float*)(ws + OFF_USAGE);
    float* wcomb = (float*)(ws + OFF_W);
    bf16*  xbf   = (bf16*)(ws + OFF_XBF);
    bf16*  W1t   = (bf16*)(ws + OFF_W1T);
    bf16*  W2t   = (bf16*)(ws + OFF_W2T);
    bf16*  hbuf  = (bf16*)(ws + OFF_H);

    float* outFinal = (float*)d_out;                                  // [T,1024]
    float* outLoss  = outFinal + (size_t)T_TOK * D_MODEL;             // scalar
    float* outTopk  = outLoss + 1;                                    // [T,2]

    init_kernel<<<1, 32, 0, stream>>>(sums, sumsq, usage);
    stats_kernel<<<dim3(64, 4), 256, 0, stream>>>(x, sums, sumsq);
    cvt_x_kernel<<<4096, 256, 0, stream>>>(x, xbf, (size_t)T_TOK * D_MODEL);
    cvt_w1_kernel<<<(N_FFN * D_MODEL + 255) / 256, 256, 0, stream>>>(W1, W1t);
    cvt_w2_kernel<<<(D_MODEL * N_FFN + 255) / 256, 256, 0, stream>>>(W2, W2t);
    gate_kernel<<<T_TOK / 16, 256, 0, stream>>>(x, task_id, task_emb, Wg, bg,
                                                sums, sumsq, wcomb, usage, outTopk);
    // GEMM1: [16384,1024] x [1024,512] -> h
    moe_gemm_kernel<1><<<dim3(N_FFN / 128, T_TOK / 128), 256, 0, stream>>>(
        xbf, W1t, wcomb, b1, hbuf, nullptr, T_TOK, N_FFN, D_MODEL);
    // GEMM2: [16384,512] x [512,1024] -> final
    moe_gemm_kernel<2><<<dim3(D_MODEL / 128, T_TOK / 128), 256, 0, stream>>>(
        hbuf, W2t, wcomb, b2, nullptr, outFinal, T_TOK, D_MODEL, N_FFN);
    loss_kernel<<<1, 32, 0, stream>>>(usage, outLoss);
}
